// GnnMapping_86758339379309
// MI455X (gfx1250) — compile-verified, matrix operand order checked by reading, not run on hardware
//
#include <hip/hip_runtime.h>

// ---------------- problem constants ----------------
#define NNODES   65536      // B * 1024
#define NEDGES   524288     // edges without self loops
#define NE2      (NEDGES + NNODES)
#define FEATC    64
#define NPG      1024
#define NGRAPH   64
#define NPAIR    65536      // 64 * 32 * 32
#define ATT_NEG  0.2f
#define ACT_NEG  0.01f

typedef __attribute__((ext_vector_type(16))) __bf16        v16bf;
typedef __attribute__((ext_vector_type(8)))  float         v8f;
typedef __attribute__((ext_vector_type(4)))  unsigned int  v4u;

__device__ __forceinline__ unsigned short f2bf(float f) {
  unsigned u = __float_as_uint(f);
  u += 0x7FFFu + ((u >> 16) & 1u);          // round-to-nearest-even
  return (unsigned short)(u >> 16);
}
__device__ __forceinline__ float lrelu(float v, float s) { return v > 0.f ? v : s * v; }

// ---------------- small utility kernels ----------------
__global__ void k_zero_i(int* p, int n) {
  int t = blockIdx.x * blockDim.x + threadIdx.x;
  if (t < n) p[t] = 0;
}
__global__ void k_copy(const float* __restrict__ a, float* __restrict__ b, int n) {
  int t = blockIdx.x * blockDim.x + threadIdx.x;
  if (t < n) b[t] = a[t];
}
// W [K, Nc] f32 -> Wt [Nc, K] bf16  (so WMMA B fragments are contiguous along K)
__global__ void k_transW(const float* __restrict__ W, unsigned short* __restrict__ Wt,
                         int K, int Nc) {
  int t = blockIdx.x * blockDim.x + threadIdx.x;
  if (t >= K * Nc) return;
  int k = t / Nc, n = t % Nc;
  Wt[n * K + k] = f2bf(W[t]);
}
// embedding gather: x = emb[idx]; writes f32 + bf16 (GEMM-ready)
__global__ void k_embed(const int* __restrict__ idx, const float* __restrict__ emb,
                        float* __restrict__ x, unsigned short* __restrict__ xbf) {
  int t = blockIdx.x * blockDim.x + threadIdx.x;
  if (t >= NNODES * FEATC) return;
  int n = t >> 6, f = t & 63;
  float v = emb[idx[n] * FEATC + f];
  x[t] = v;
  xbf[t] = f2bf(v);
}

// ---------------- CSR construction (edges fixed for all 12 layers) ----------------
__global__ void k_degree(const int* __restrict__ dstE, int* __restrict__ deg) {
  int e = blockIdx.x * blockDim.x + threadIdx.x;
  if (e >= NE2) return;
  int d = e < NEDGES ? dstE[e] : e - NEDGES;
  atomicAdd(&deg[d], 1);
}
// block-level inclusive scan -> exclusive rowstart + block totals (N = 256*256 exactly)
__global__ void k_scan_block(const int* __restrict__ deg, int* __restrict__ rowstart,
                             int* __restrict__ bsum) {
  __shared__ int sh[256];
  int i = blockIdx.x * 256 + threadIdx.x;
  int v = deg[i];
  sh[threadIdx.x] = v;
  __syncthreads();
  for (int st = 1; st < 256; st <<= 1) {
    int t = (threadIdx.x >= st) ? sh[threadIdx.x - st] : 0;
    __syncthreads();
    sh[threadIdx.x] += t;
    __syncthreads();
  }
  rowstart[i] = sh[threadIdx.x] - v;                 // exclusive
  if (threadIdx.x == 255) bsum[blockIdx.x] = sh[255];
}
__global__ void k_scan_top(int* __restrict__ bsum) {  // exclusive scan of 256 values
  __shared__ int sh[256];
  int v = bsum[threadIdx.x];
  sh[threadIdx.x] = v;
  __syncthreads();
  for (int st = 1; st < 256; st <<= 1) {
    int t = (threadIdx.x >= st) ? sh[threadIdx.x - st] : 0;
    __syncthreads();
    sh[threadIdx.x] += t;
    __syncthreads();
  }
  bsum[threadIdx.x] = sh[threadIdx.x] - v;
}
__global__ void k_scan_add(int* __restrict__ rowstart, const int* __restrict__ bsum) {
  int i = blockIdx.x * blockDim.x + threadIdx.x;
  if (i < NNODES) rowstart[i] += bsum[i >> 8];
}
__global__ void k_fill_csr(const int* __restrict__ srcE, const int* __restrict__ dstE,
                           const int* __restrict__ rowstart, int* __restrict__ cursor,
                           int* __restrict__ csrc) {
  int e = blockIdx.x * blockDim.x + threadIdx.x;
  if (e >= NE2) return;
  int s = e < NEDGES ? srcE[e] : e - NEDGES;
  int d = e < NEDGES ? dstE[e] : e - NEDGES;
  int slot = rowstart[d] + atomicAdd(&cursor[d], 1);
  csrc[slot] = s;
}

// ---------------- WMMA GEMM: C[M,Nc] = A[M,K] * Wt^T, bf16 in / f32 acc ----------------
template <int K>
__global__ void k_wmma_gemm(const unsigned short* __restrict__ A,   // [M,K] bf16 row-major
                            const unsigned short* __restrict__ Wt,  // [Nc,K] bf16 row-major
                            float* __restrict__ C,                  // [M,Nc] f32
                            int M, int Nc) {
  constexpr int KS = K / 32;
  const int lane = threadIdx.x & 31;
  const int wave = threadIdx.x >> 5;
  const int m0   = (blockIdx.x * 8 + wave) * 16;
  if (m0 >= M) return;
  const int rsub = lane & 15;            // row (A) / col (B) within tile
  const int kb   = (lane >> 4) * 8;      // K sub-base per ISA 16-bit fragment layout

  union Frag { v4u q[2]; v16bf v; };
  Frag a[KS];
  const unsigned short* ap = A + (size_t)(m0 + rsub) * K + kb;
#pragma unroll
  for (int ks = 0; ks < KS; ++ks) {
    a[ks].q[0] = *(const v4u*)(ap + ks * 32);        // K = k0+kb .. +7
    a[ks].q[1] = *(const v4u*)(ap + ks * 32 + 16);   // K = k0+16+kb .. +7
  }
  for (int n0 = 0; n0 < Nc; n0 += 16) {
    const int col = n0 + rsub;
    const unsigned short* bp = Wt + (size_t)col * K + kb;
    v8f acc = {};
#pragma unroll
    for (int ks = 0; ks < KS; ++ks) {
      Frag b;
      b.q[0] = *(const v4u*)(bp + ks * 32);
      b.q[1] = *(const v4u*)(bp + ks * 32 + 16);
      acc = __builtin_amdgcn_wmma_f32_16x16x32_bf16(
          false, a[ks].v, false, b.v, (short)0, acc, false, false);
    }
    const int mrow = m0 + (lane >> 4) * 8;
#pragma unroll
    for (int i = 0; i < 8; ++i)
      C[(size_t)(mrow + i) * Nc + col] = acc[i];
  }
}

// ---------------- GAT attention (CSR, atomic-free) ----------------
// per (node, head): a_s = <h, a_src>, a_d = <h, a_dst>
__global__ void k_attn_coef(const float* __restrict__ h, const float* __restrict__ asrc,
                            const float* __restrict__ adst, float* __restrict__ as_,
                            float* __restrict__ ad_, int H) {
  int t = blockIdx.x * blockDim.x + threadIdx.x;
  if (t >= NNODES * H) return;
  int n = t / H, hd = t % H;
  const float* hr = h + (size_t)n * H * FEATC + hd * FEATC;
  float s1 = 0.f, s2 = 0.f;
#pragma unroll 8
  for (int c = 0; c < FEATC; ++c) {
    float v = hr[c];
    s1 += v * asrc[hd * FEATC + c];
    s2 += v * adst[hd * FEATC + c];
  }
  as_[t] = s1;
  ad_[t] = s2;
}
// per (dst node, head): softmax over in-edges, alpha stored per CSR slot
__global__ void k_node_softmax(const int* __restrict__ rowstart, const int* __restrict__ deg,
                               const int* __restrict__ csrc, const float* __restrict__ as_,
                               const float* __restrict__ ad_, float* __restrict__ alpha,
                               int H) {
  int t = blockIdx.x * blockDim.x + threadIdx.x;
  if (t >= NNODES * H) return;
  int n = t / H, hd = t % H;
  int r0 = rowstart[n], dg = deg[n];
  float adv = ad_[n * H + hd];
  float m = -3.0e38f;
  for (int k = 0; k < dg; ++k) {
    float v = lrelu(as_[csrc[r0 + k] * H + hd] + adv, ATT_NEG);
    m = fmaxf(m, v);
  }
  float den = 0.f;
  for (int k = 0; k < dg; ++k) {
    float v = lrelu(as_[csrc[r0 + k] * H + hd] + adv, ATT_NEG);
    float ex = __expf(v - m);
    alpha[(size_t)(r0 + k) * H + hd] = ex;
    den += ex;
  }
  float inv = 1.f / (den + 1e-16f);
  for (int k = 0; k < dg; ++k)
    alpha[(size_t)(r0 + k) * H + hd] *= inv;
}
// block per dst node, thread per feature: y = lrelu(sum_k h[src_k]*alpha_k + bias)(+resid)
// writes f32 + bf16 (next GEMM input) in one pass; no atomics anywhere.
__global__ void k_node_aggr(const int* __restrict__ rowstart, const int* __restrict__ deg,
                            const int* __restrict__ csrc, const float* __restrict__ h,
                            const float* __restrict__ alpha, const float* __restrict__ bias,
                            const float* __restrict__ resid, float* __restrict__ y,
                            unsigned short* __restrict__ ybf, int H, int addres) {
  int n = blockIdx.x;
  int f = threadIdx.x;                  // [0, H*64)
  int HC = H * FEATC;
  int hd = f >> 6;
  int r0 = rowstart[n], dg = deg[n];
  float acc = 0.f;
  for (int k = 0; k < dg; ++k) {
    int s = csrc[r0 + k];
    if (k + 1 < dg)                      // gfx1250 global_prefetch of next src row
      __builtin_prefetch(&h[(size_t)csrc[r0 + k + 1] * HC + f], 0, 1);
    acc += h[(size_t)s * HC + f] * alpha[(size_t)(r0 + k) * H + hd];
  }
  float v = lrelu(acc + bias[f], ACT_NEG);
  if (addres) v += resid[(size_t)n * HC + f];
  size_t idx = (size_t)n * HC + f;
  y[idx] = v;
  ybf[idx] = f2bf(v);
}

// ---------------- graph LayerNorm (PyG mode='graph'), in place; emits bf16 too ----------
__global__ void k_ln_graph(float* __restrict__ y, unsigned short* __restrict__ ybf,
                           const float* __restrict__ w, const float* __restrict__ b) {
  int g = blockIdx.x;
  float* yg = y + (size_t)g * NPG * FEATC;
  unsigned short* yb = ybf + (size_t)g * NPG * FEATC;
  const int total = NPG * FEATC;
  float s = 0.f, s2 = 0.f;
  for (int i = threadIdx.x; i < total; i += blockDim.x) {
    float v = yg[i]; s += v; s2 += v * v;
  }
  __shared__ float r1[256], r2[256];
  r1[threadIdx.x] = s; r2[threadIdx.x] = s2;
  __syncthreads();
  for (int st = 128; st > 0; st >>= 1) {
    if (threadIdx.x < st) { r1[threadIdx.x] += r1[threadIdx.x + st]; r2[threadIdx.x] += r2[threadIdx.x + st]; }
    __syncthreads();
  }
  float mean = r1[0] / total;
  float var  = r2[0] / total - mean * mean;
  float inv  = rsqrtf(var + 1e-5f);
  for (int i = threadIdx.x; i < total; i += blockDim.x) {
    int c = i & (FEATC - 1);
    float v = (yg[i] - mean) * inv * w[c] + b[c];
    yg[i] = v;
    yb[i] = f2bf(v);
  }
}

// ---------------- pairwise MLP helpers ----------------
__global__ void k_build_pair(const float* __restrict__ x, unsigned short* __restrict__ pbf) {
  int t = blockIdx.x * blockDim.x + threadIdx.x;   // over NPAIR*128
  if (t >= NPAIR * 128) return;
  int p = t >> 7, f = t & 127;
  int b = p >> 10, i = (p >> 5) & 31, j = p & 31;
  int node = b * NPG + ((f < 64) ? i : j);
  pbf[t] = f2bf(x[(size_t)node * FEATC + (f & 63)]);
}
__global__ void k_bias_act(const float* __restrict__ agg, const float* __restrict__ bias,
                           float* __restrict__ y, unsigned short* __restrict__ ybf,
                           int total, int Wd) {
  int t = blockIdx.x * blockDim.x + threadIdx.x;
  if (t >= total) return;
  float v = lrelu(agg[t] + bias[t % Wd], ACT_NEG);
  y[t] = v;
  ybf[t] = f2bf(v);
}
__global__ void k_mlp_final(const float* __restrict__ h, const float* __restrict__ W,
                            const float* __restrict__ bias, float* __restrict__ s) {
  int p = blockIdx.x * blockDim.x + threadIdx.x;
  if (p >= NPAIR) return;
  float acc = bias[0];
#pragma unroll 8
  for (int c = 0; c < 64; ++c) acc += h[(size_t)p * 64 + c] * W[c];
  s[p] = acc;
}
__global__ void k_symmetrize(const float* __restrict__ s, float* __restrict__ out) {
  int t = blockIdx.x * blockDim.x + threadIdx.x;
  if (t >= NPAIR) return;
  int b = t >> 10, i = (t >> 5) & 31, j = t & 31;
  out[t] = 0.5f * (s[(b * 32 + i) * 32 + j] + s[(b * 32 + j) * 32 + i]);
}

// ---------------- host orchestration ----------------
struct GatP { const float *W, *a_dst, *a_src, *b; };

extern "C" void kernel_launch(void* const* d_in, const int* in_sizes, int n_in,
                              void* d_out, int out_size, void* d_ws, size_t ws_size,
                              hipStream_t stream) {
  const int* xidx = (const int*)d_in[0];
  const int* eidx = (const int*)d_in[1];
  const int* srcE = eidx;
  const int* dstE = eidx + NEDGES;
  // params flattened in jax pytree order (dict keys sorted):
  // blocks[2]{ layers[6]{W,a_dst,a_src,b}, ln_b, ln_w }, emb, mlp[4]{W,b}
  int pi = 3;
  GatP blk[2][6]; const float *lnb[2], *lnw[2];
  for (int bi = 0; bi < 2; ++bi) {
    for (int l = 0; l < 6; ++l) {
      blk[bi][l].W     = (const float*)d_in[pi++];
      blk[bi][l].a_dst = (const float*)d_in[pi++];
      blk[bi][l].a_src = (const float*)d_in[pi++];
      blk[bi][l].b     = (const float*)d_in[pi++];
    }
    lnb[bi] = (const float*)d_in[pi++];
    lnw[bi] = (const float*)d_in[pi++];
  }
  const float* emb = (const float*)d_in[pi++];
  const float *mW[4], *mB[4];
  for (int i = 0; i < 4; ++i) { mW[i] = (const float*)d_in[pi++]; mB[i] = (const float*)d_in[pi++]; }

  // workspace layout
  char* w = (char*)d_ws;
  size_t o = 0;
  float* xbuf   = (float*)(w + o); o += (size_t)NNODES * 128 * 4;
  float* hbuf   = (float*)(w + o); o += (size_t)NNODES * 128 * 4;
  float* aggbuf = (float*)(w + o); o += (size_t)NNODES * 128 * 4;   // MLP GEMM out
  float* resid  = (float*)(w + o); o += (size_t)NNODES * 64 * 4;
  float* as_    = (float*)(w + o); o += (size_t)NNODES * 2 * 4;
  float* ad_    = (float*)(w + o); o += (size_t)NNODES * 2 * 4;
  float* alpha  = (float*)(w + o); o += (size_t)NE2 * 2 * 4;
  float* sbuf   = (float*)(w + o); o += (size_t)NPAIR * 4;
  int* deg      = (int*)(w + o);   o += (size_t)NNODES * 4;
  int* rowstart = (int*)(w + o);   o += (size_t)NNODES * 4;
  int* cursor   = (int*)(w + o);   o += (size_t)NNODES * 4;
  int* bsum     = (int*)(w + o);   o += 256 * 4;
  int* csrc     = (int*)(w + o);   o += (size_t)NE2 * 4;
  o = (o + 255) & ~(size_t)255;
  unsigned short* abf = (unsigned short*)(w + o); o += (size_t)NNODES * 128 * 2;
  unsigned short* wtb = (unsigned short*)(w + o); o += 128 * 128 * 2;

  auto cdiv = [](int a, int b) { return (a + b - 1) / b; };

  // ---- build CSR by dst once; reused by all 12 GAT layers ----
  k_zero_i<<<cdiv(NNODES, 256), 256, 0, stream>>>(deg, NNODES);
  k_zero_i<<<cdiv(NNODES, 256), 256, 0, stream>>>(cursor, NNODES);
  k_degree<<<cdiv(NE2, 256), 256, 0, stream>>>(dstE, deg);
  k_scan_block<<<256, 256, 0, stream>>>(deg, rowstart, bsum);
  k_scan_top<<<1, 256, 0, stream>>>(bsum);
  k_scan_add<<<cdiv(NNODES, 256), 256, 0, stream>>>(rowstart, bsum);
  k_fill_csr<<<cdiv(NE2, 256), 256, 0, stream>>>(srcE, dstE, rowstart, cursor, csrc);

  // GEMM on pre-converted bf16 activations
  auto gemm = [&](const unsigned short* Abf, const float* Wf32, float* Cout,
                  int M, int inW, int Nc) {
    k_transW<<<cdiv(inW * Nc, 256), 256, 0, stream>>>(Wf32, wtb, inW, Nc);
    if (inW == 64)
      k_wmma_gemm<64><<<M / 128, 256, 0, stream>>>(Abf, wtb, Cout, M, Nc);
    else
      k_wmma_gemm<128><<<M / 128, 256, 0, stream>>>(Abf, wtb, Cout, M, Nc);
  };

  auto run_gat = [&](int inW, const GatP& p, int H, int addres) {
    const int HC = H * FEATC;
    gemm(abf, p.W, hbuf, NNODES, inW, HC);                         // h = x @ W
    k_attn_coef<<<cdiv(NNODES * H, 256), 256, 0, stream>>>(hbuf, p.a_src, p.a_dst,
                                                           as_, ad_, H);
    k_node_softmax<<<cdiv(NNODES * H, 256), 256, 0, stream>>>(rowstart, deg, csrc,
                                                              as_, ad_, alpha, H);
    k_node_aggr<<<NNODES, HC, 0, stream>>>(rowstart, deg, csrc, hbuf, alpha,
                                           p.b, resid, xbuf, abf, H, addres);
  };

  // x = emb[x_idx]  (f32 + bf16)
  k_embed<<<cdiv(NNODES * FEATC, 256), 256, 0, stream>>>(xidx, emb, xbuf, abf);

  for (int bi = 0; bi < 2; ++bi) {
    k_copy<<<cdiv(NNODES * 64, 256), 256, 0, stream>>>(xbuf, resid, NNODES * 64);
    run_gat(64,  blk[bi][0], 2, 0);    // 64  -> 128
    run_gat(128, blk[bi][1], 2, 0);    // 128 -> 128
    run_gat(128, blk[bi][2], 2, 0);
    run_gat(128, blk[bi][3], 2, 0);
    run_gat(128, blk[bi][4], 2, 0);
    run_gat(128, blk[bi][5], 1, 1);    // 128 -> 64, + residual
    k_ln_graph<<<NGRAPH, 256, 0, stream>>>(xbuf, abf, lnw[bi], lnb[bi]);
  }

  // pairwise MLP head: pair [NPAIR,128] -> 128 -> 64 -> 64 -> 1
  k_build_pair<<<cdiv(NPAIR * 128, 256), 256, 0, stream>>>(xbuf, abf);
  gemm(abf, mW[0], aggbuf, NPAIR, 128, 128);
  k_bias_act<<<cdiv(NPAIR * 128, 256), 256, 0, stream>>>(aggbuf, mB[0], hbuf, abf,
                                                         NPAIR * 128, 128);
  gemm(abf, mW[1], aggbuf, NPAIR, 128, 64);
  k_bias_act<<<cdiv(NPAIR * 64, 256), 256, 0, stream>>>(aggbuf, mB[1], hbuf, abf,
                                                        NPAIR * 64, 64);
  gemm(abf, mW[2], aggbuf, NPAIR, 64, 64);
  k_bias_act<<<cdiv(NPAIR * 64, 256), 256, 0, stream>>>(aggbuf, mB[2], hbuf, abf,
                                                        NPAIR * 64, 64);
  k_mlp_final<<<cdiv(NPAIR, 256), 256, 0, stream>>>(hbuf, mW[3], mB[3], sbuf);
  k_symmetrize<<<cdiv(NPAIR, 256), 256, 0, stream>>>(sbuf, (float*)d_out);
}
